// Decoder_7962869367391
// MI455X (gfx1250) — matrix operand order repaired and checked by reading, not compiled, not run on hardware
//
#include <hip/hip_runtime.h>

#define B_ 64
#define S_ 400
#define L_ 100
#define E_ 512
#define H_ 512

typedef __attribute__((ext_vector_type(16))) __bf16 v16bf;
typedef __attribute__((ext_vector_type(8)))  __bf16 v8bf;
typedef __attribute__((ext_vector_type(8)))  float  v8f;

__device__ __forceinline__ __bf16 to_bf(float f) { return (__bf16)f; }

__device__ __forceinline__ v16bf cat16(v8bf lo, v8bf hi) {
    return __builtin_shufflevector(lo, hi, 0,1,2,3,4,5,6,7,8,9,10,11,12,13,14,15);
}

// Native v_tanh_f32 if the toolchain exposes it, else libm tanhf.
__device__ __forceinline__ float fast_tanh(float x) {
#if __has_builtin(__builtin_amdgcn_tanhf)
    return __builtin_amdgcn_tanhf(x);
#elif __has_builtin(__builtin_amdgcn_tanh_f32)
    return __builtin_amdgcn_tanh_f32(x);
#else
    return tanhf(x);
#endif
}

// ---------------------------------------------------------------------------
// One-time GEMM (f32 inputs, convert in-register). Bridge + proj_key only.
// ---------------------------------------------------------------------------
__global__ void gemm_f32in(const float* __restrict__ A, int lda,
                           const float* __restrict__ W, int ldw,
                           const float* __restrict__ bias,
                           float* __restrict__ out, long ldo,
                           int M, int N, int K, int act)
{
    const int lane = threadIdx.x & 31;
    const int wave = threadIdx.x >> 5;
    const int mtiles = M >> 4;
    const int ntiles = N >> 4;
    const int tile = blockIdx.x * 8 + wave;
    if (tile >= mtiles * ntiles) return;
    const int mt  = tile % mtiles;
    const int nt  = tile / mtiles;
    const int grp = lane >> 4;
    const int l15 = lane & 15;

    const float* arow = A + (size_t)(mt * 16 + l15) * lda + grp * 8;
    const float* wrow = W + (size_t)(nt * 16 + l15) * ldw + grp * 8;

    v8f acc = {0.f, 0.f, 0.f, 0.f, 0.f, 0.f, 0.f, 0.f};

    for (int k0 = 0; k0 < K; k0 += 32) {
        const float* pa = arow + k0;
        float4 a0 = *(const float4*)(pa);
        float4 a1 = *(const float4*)(pa + 4);
        float4 a2 = *(const float4*)(pa + 16);
        float4 a3 = *(const float4*)(pa + 20);
        const float* pw = wrow + k0;
        float4 b0 = *(const float4*)(pw);
        float4 b1 = *(const float4*)(pw + 4);
        float4 b2 = *(const float4*)(pw + 16);
        float4 b3 = *(const float4*)(pw + 20);

        v16bf fa, fb;
        fa[0]  = to_bf(a0.x); fa[1]  = to_bf(a0.y); fa[2]  = to_bf(a0.z); fa[3]  = to_bf(a0.w);
        fa[4]  = to_bf(a1.x); fa[5]  = to_bf(a1.y); fa[6]  = to_bf(a1.z); fa[7]  = to_bf(a1.w);
        fa[8]  = to_bf(a2.x); fa[9]  = to_bf(a2.y); fa[10] = to_bf(a2.z); fa[11] = to_bf(a2.w);
        fa[12] = to_bf(a3.x); fa[13] = to_bf(a3.y); fa[14] = to_bf(a3.z); fa[15] = to_bf(a3.w);
        fb[0]  = to_bf(b0.x); fb[1]  = to_bf(b0.y); fb[2]  = to_bf(b0.z); fb[3]  = to_bf(b0.w);
        fb[4]  = to_bf(b1.x); fb[5]  = to_bf(b1.y); fb[6]  = to_bf(b1.z); fb[7]  = to_bf(b1.w);
        fb[8]  = to_bf(b2.x); fb[9]  = to_bf(b2.y); fb[10] = to_bf(b2.z); fb[11] = to_bf(b2.w);
        fb[12] = to_bf(b3.x); fb[13] = to_bf(b3.y); fb[14] = to_bf(b3.z); fb[15] = to_bf(b3.w);

        acc = __builtin_amdgcn_wmma_f32_16x16x32_bf16(
            false, fa, false, fb, (short)0, acc, false, false);
    }

    const float bv = bias ? bias[nt * 16 + l15] : 0.f;
    #pragma unroll
    for (int j = 0; j < 8; ++j) {
        const int m = mt * 16 + j + grp * 8;
        const int n = nt * 16 + l15;
        float v = acc[j] + bv;
        if (act == 1) v = fast_tanh(v);
        out[(size_t)m * (size_t)ldo + n] = v;
    }
}

// ---------------------------------------------------------------------------
// Steady-state GEMM: bf16 A x bf16 W^T -> f32. One wave -> 16x64 (4 n-tiles).
// ---------------------------------------------------------------------------
__global__ void gemm_bf16(const __bf16* __restrict__ A, int lda,
                          const __bf16* __restrict__ W, int ldw,
                          const float* __restrict__ bias,
                          float* __restrict__ out, long ldo,
                          int M, int N, int K)
{
    const int lane = threadIdx.x & 31;
    const int wave = threadIdx.x >> 5;
    const int mtiles  = M >> 4;
    const int ngroups = N >> 6;
    const int g = blockIdx.x * 8 + wave;
    if (g >= mtiles * ngroups) return;
    const int mt  = g % mtiles;
    const int ng  = g / mtiles;
    const int grp = lane >> 4;
    const int l15 = lane & 15;

    const __bf16* arow  = A + (size_t)(mt * 16 + l15) * lda + grp * 8;
    const __bf16* wbase = W + (size_t)(ng * 64 + l15) * ldw + grp * 8;

    const v8f zero = {0.f, 0.f, 0.f, 0.f, 0.f, 0.f, 0.f, 0.f};
    v8f acc[4] = {zero, zero, zero, zero};

    for (int k0 = 0; k0 < K; k0 += 32) {
        v8bf alo = *(const v8bf*)(arow + k0);
        v8bf ahi = *(const v8bf*)(arow + k0 + 16);
        v16bf fa = cat16(alo, ahi);
        #pragma unroll
        for (int t = 0; t < 4; ++t) {
            const __bf16* wr = wbase + (size_t)t * 16 * ldw + k0;
            v8bf blo = *(const v8bf*)(wr);
            v8bf bhi = *(const v8bf*)(wr + 16);
            v16bf fb = cat16(blo, bhi);
            acc[t] = __builtin_amdgcn_wmma_f32_16x16x32_bf16(
                false, fa, false, fb, (short)0, acc[t], false, false);
        }
    }

    #pragma unroll
    for (int t = 0; t < 4; ++t) {
        const int n = ng * 64 + t * 16 + l15;
        const float bv = bias ? bias[n] : 0.f;
        #pragma unroll
        for (int j = 0; j < 8; ++j) {
            const int m = mt * 16 + j + grp * 8;
            out[(size_t)m * (size_t)ldo + n] = acc[t][j] + bv;
        }
    }
}

__global__ void cvt_bf16(const float* __restrict__ in, __bf16* __restrict__ out, long n)
{
    long i = (long)blockIdx.x * 256 + threadIdx.x;
    if (i < n) out[i] = to_bf(in[i]);
}

// ---------------------------------------------------------------------------
// Attention pass 1: scores[b,s] = mask ? v . tanh(q[b]+proj_key[b,s]) : -1e9
// Grid: B * (S/16) blocks; 16 s-rows per block, 16 threads per row.
// ---------------------------------------------------------------------------
__global__ void scores_kernel(const float* __restrict__ q,
                              const float* __restrict__ proj_key,
                              const unsigned char* __restrict__ mask,
                              const float* __restrict__ ve,
                              float* __restrict__ scores)
{
    __shared__ float qs[H_];
    __shared__ float vs[H_];
    __shared__ float red[256];
    const int nchunk = S_ / 16;
    const int b   = blockIdx.x / nchunk;
    const int s0  = (blockIdx.x % nchunk) * 16;
    const int tid = threadIdx.x;
    for (int j = tid; j < H_; j += 256) { qs[j] = q[b * H_ + j]; vs[j] = ve[j]; }
    __syncthreads();

    const int r = tid >> 4;          // row in chunk (0..15)
    const int c = tid & 15;          // partial-sum lane (0..15)
    const int s = s0 + r;
    const float* pk = proj_key + ((size_t)b * S_ + s) * H_;
    float acc = 0.f;
    for (int j = c * 4; j < H_; j += 64) {
        float4 p = *(const float4*)(pk + j);
        acc += vs[j    ] * fast_tanh(qs[j    ] + p.x);
        acc += vs[j + 1] * fast_tanh(qs[j + 1] + p.y);
        acc += vs[j + 2] * fast_tanh(qs[j + 2] + p.z);
        acc += vs[j + 3] * fast_tanh(qs[j + 3] + p.w);
    }
    red[tid] = acc;
    __syncthreads();
    for (int off = 8; off > 0; off >>= 1) {
        if (c < off) red[tid] += red[tid + off];
        __syncthreads();
    }
    if (c == 0) {
        float v = red[r * 16];
        scores[(size_t)b * S_ + s] = mask[b * S_ + s] ? v : -1e9f;
    }
}

// Attention pass 2: in-place masked softmax over S. One block per batch row.
__global__ void softmax_kernel(float* __restrict__ scores)
{
    __shared__ float sc[S_];
    __shared__ float red[256];
    const int b = blockIdx.x;
    const int tid = threadIdx.x;
    float* sb = scores + (size_t)b * S_;
    for (int s = tid; s < S_; s += 256) sc[s] = sb[s];
    __syncthreads();

    float m = -3.0e38f;
    for (int s = tid; s < S_; s += 256) m = fmaxf(m, sc[s]);
    red[tid] = m; __syncthreads();
    for (int off = 128; off > 0; off >>= 1) {
        if (tid < off) red[tid] = fmaxf(red[tid], red[tid + off]);
        __syncthreads();
    }
    const float mx = red[0]; __syncthreads();
    float sum = 0.f;
    for (int s = tid; s < S_; s += 256) { float e = __expf(sc[s] - mx); sc[s] = e; sum += e; }
    red[tid] = sum; __syncthreads();
    for (int off = 128; off > 0; off >>= 1) {
        if (tid < off) red[tid] += red[tid + off];
        __syncthreads();
    }
    const float inv = 1.f / red[0];
    for (int s = tid; s < S_; s += 256) sb[s] = sc[s] * inv;
}

// Attention pass 3: context[b,d] = sum_s alpha[b,s] * enc[b,s,d].
// Grid: B*4 blocks (256 dims each), alphas staged in LDS.
__global__ void ctx_kernel(const float* __restrict__ alphas,
                           const float* __restrict__ enc,
                           float* __restrict__ ctx)
{
    __shared__ float al[S_];
    const int b = blockIdx.x >> 2;
    const int d = (blockIdx.x & 3) * 256 + threadIdx.x;
    for (int s = threadIdx.x; s < S_; s += 256) al[s] = alphas[(size_t)b * S_ + s];
    __syncthreads();
    const float* eb = enc + (size_t)b * S_ * (2 * H_) + d;
    float acc = 0.f;
    for (int s = 0; s < S_; ++s) acc += al[s] * eb[(size_t)s * (2 * H_)];
    ctx[b * (2 * H_) + d] = acc;
}

// x_bf16 = [prev_embed, context]
__global__ void pack_x(const float* __restrict__ trg, const float* __restrict__ sos,
                       const float* __restrict__ ctx, __bf16* __restrict__ x, int step)
{
    const int W = E_ + 2 * H_;
    int idx = blockIdx.x * 256 + threadIdx.x;
    if (idx >= B_ * W) return;
    int b = idx / W, c = idx % W;
    float v;
    if (c < E_) v = (step == 0) ? sos[c] : trg[((size_t)b * L_ + (step - 1)) * E_ + c];
    else        v = ctx[b * (2 * H_) + (c - E_)];
    x[idx] = to_bf(v);
}

// GRU gates: updates h (f32), h_bf16, writes decoder_states
__global__ void gate_kernel(const float* __restrict__ gi, const float* __restrict__ gh,
                            float* __restrict__ h, __bf16* __restrict__ hbf,
                            float* __restrict__ states, int step)
{
    int idx = blockIdx.x * 256 + threadIdx.x;
    if (idx >= B_ * H_) return;
    int b = idx / H_, j = idx % H_;
    const float* gib = gi + (size_t)b * 3 * H_;
    const float* ghb = gh + (size_t)b * 3 * H_;
    float ir = gib[j], iz = gib[H_ + j], in_ = gib[2 * H_ + j];
    float hr = ghb[j], hz = ghb[H_ + j], hn  = ghb[2 * H_ + j];
    float r = 1.f / (1.f + __expf(-(ir + hr)));
    float z = 1.f / (1.f + __expf(-(iz + hz)));
    float n = fast_tanh(in_ + r * hn);
    float hnew = (1.f - z) * n + z * h[idx];
    h[idx]   = hnew;
    hbf[idx] = to_bf(hnew);
    states[((size_t)b * L_ + step) * H_ + j] = hnew;
}

// prein_bf16 = [prev_embed, h_new, context]
__global__ void pack_pre(const float* __restrict__ trg, const float* __restrict__ sos,
                         const float* __restrict__ h, const float* __restrict__ ctx,
                         __bf16* __restrict__ prein, int step)
{
    const int W = E_ + H_ + 2 * H_;
    int idx = blockIdx.x * 256 + threadIdx.x;
    if (idx >= B_ * W) return;
    int b = idx / W, c = idx % W;
    float v;
    if (c < E_)            v = (step == 0) ? sos[c] : trg[((size_t)b * L_ + (step - 1)) * E_ + c];
    else if (c < E_ + H_)  v = h[b * H_ + (c - E_)];
    else                   v = ctx[b * (2 * H_) + (c - E_ - H_)];
    prein[idx] = to_bf(v);
}

__global__ void copy_h(const float* __restrict__ h, float* __restrict__ hf)
{
    int idx = blockIdx.x * 256 + threadIdx.x;
    if (idx < B_ * H_) hf[idx] = h[idx];
}

extern "C" void kernel_launch(void* const* d_in, const int* in_sizes, int n_in,
                              void* d_out, int out_size, void* d_ws, size_t ws_size,
                              hipStream_t stream)
{
    const float* trg   = (const float*)d_in[0];
    const float* enc   = (const float*)d_in[1];
    const float* encf  = (const float*)d_in[2];
    const unsigned char* mask = (const unsigned char*)d_in[3];
    const float* Wk    = (const float*)d_in[5];
    const float* bk    = (const float*)d_in[6];
    const float* Wq    = (const float*)d_in[7];
    const float* bq    = (const float*)d_in[8];
    const float* v_e   = (const float*)d_in[9];
    const float* Wbr   = (const float*)d_in[10];
    const float* bbr   = (const float*)d_in[11];
    const float* W_ih  = (const float*)d_in[12];
    const float* b_ih  = (const float*)d_in[13];
    const float* W_hh  = (const float*)d_in[14];
    const float* b_hh  = (const float*)d_in[15];
    const float* W_pre = (const float*)d_in[16];
    const float* sos   = (const float*)d_in[17];

    // ---- workspace: f32 region, then bf16 region ----
    float* ws       = (float*)d_ws;
    float* proj_key = ws;                                // B*S*H
    float* h        = proj_key + (size_t)B_ * S_ * H_;   // B*H
    float* q        = h      + (size_t)B_ * H_;          // B*H
    float* ctx      = q      + (size_t)B_ * H_;          // B*2H
    float* gi       = ctx    + (size_t)B_ * 2 * H_;      // B*3H
    float* gh       = gi     + (size_t)B_ * 3 * H_;      // B*3H
    float* scores   = gh     + (size_t)B_ * 3 * H_;      // B*S
    float* f32_end  = scores + (size_t)B_ * S_;

    __bf16* Wq_bf   = (__bf16*)f32_end;                      // H*H
    __bf16* Wih_bf  = Wq_bf  + (size_t)H_ * H_;              // 3H*(E+2H)
    __bf16* Whh_bf  = Wih_bf + (size_t)3 * H_ * (E_ + 2*H_); // 3H*H
    __bf16* Wpre_bf = Whh_bf + (size_t)3 * H_ * H_;          // H*(E+3H)
    __bf16* h_bf    = Wpre_bf+ (size_t)H_ * (E_ + 3*H_);     // B*H
    __bf16* x_bf    = h_bf   + (size_t)B_ * H_;              // B*(E+2H)
    __bf16* pre_bf  = x_bf   + (size_t)B_ * (E_ + 2*H_);     // B*(E+3H)

    float* states = (float*)d_out;
    float* hfinal = states + (size_t)B_ * L_ * H_;
    float* preout = hfinal + (size_t)B_ * H_;

    auto gemmF = [&](const float* A, int lda, const float* W, int ldw,
                     const float* bias, float* O, long ldo,
                     int M, int N, int K, int act) {
        int tiles = (M / 16) * (N / 16);
        gemm_f32in<<<(tiles + 7) / 8, 256, 0, stream>>>(A, lda, W, ldw, bias, O, ldo, M, N, K, act);
    };
    auto gemmB = [&](const __bf16* A, int lda, const __bf16* W, int ldw,
                     const float* bias, float* O, long ldo,
                     int M, int N, int K) {
        int groups = (M / 16) * (N / 64);
        gemm_bf16<<<(groups + 7) / 8, 256, 0, stream>>>(A, lda, W, ldw, bias, O, ldo, M, N, K);
    };
    auto cvt = [&](const float* in, __bf16* out, long n) {
        cvt_bf16<<<(int)((n + 255) / 256), 256, 0, stream>>>(in, out, n);
    };

    // one-time weight conversions (L2-resident thereafter)
    cvt(Wq,    Wq_bf,   (long)H_ * H_);
    cvt(W_ih,  Wih_bf,  (long)3 * H_ * (E_ + 2 * H_));
    cvt(W_hh,  Whh_bf,  (long)3 * H_ * H_);
    cvt(W_pre, Wpre_bf, (long)H_ * (E_ + 3 * H_));

    // h0 = tanh(encoder_final[0] @ Wbr^T + bbr)
    gemmF(encf, 2 * H_, Wbr, 2 * H_, bbr, h, H_, B_, H_, 2 * H_, 1);
    cvt(h, h_bf, (long)B_ * H_);
    // proj_key = enc @ Wk^T + bk
    gemmF(enc, 2 * H_, Wk, 2 * H_, bk, proj_key, H_, B_ * S_, H_, 2 * H_, 0);

    for (int i = 0; i < L_; ++i) {
        // q = h @ Wq^T + bq
        gemmB(h_bf, H_, Wq_bf, H_, bq, q, H_, B_, H_, H_);
        // attention (3 passes, parallel across the WGP array)
        scores_kernel<<<B_ * (S_ / 16), 256, 0, stream>>>(q, proj_key, mask, v_e, scores);
        softmax_kernel<<<B_, 256, 0, stream>>>(scores);
        ctx_kernel<<<B_ * 4, 256, 0, stream>>>(scores, enc, ctx);
        pack_x<<<(B_ * (E_ + 2 * H_) + 255) / 256, 256, 0, stream>>>(trg, sos, ctx, x_bf, i);
        // gi = x @ W_ih^T + b_ih ; gh = h @ W_hh^T + b_hh
        gemmB(x_bf, E_ + 2 * H_, Wih_bf, E_ + 2 * H_, b_ih, gi, 3 * H_, B_, 3 * H_, E_ + 2 * H_);
        gemmB(h_bf, H_, Whh_bf, H_, b_hh, gh, 3 * H_, B_, 3 * H_, H_);
        gate_kernel<<<(B_ * H_ + 255) / 256, 256, 0, stream>>>(gi, gh, h, h_bf, states, i);
        pack_pre<<<(B_ * (E_ + 3 * H_) + 255) / 256, 256, 0, stream>>>(trg, sos, h, ctx, pre_bf, i);
        // pre_output[:, i, :] = pre_in @ W_pre^T  (strided store into output)
        gemmB(pre_bf, E_ + 3 * H_, Wpre_bf, E_ + 3 * H_, nullptr,
              preout + (size_t)i * H_, (long)L_ * H_, B_, H_, E_ + 3 * H_);
    }

    copy_h<<<(B_ * H_ + 255) / 256, 256, 0, stream>>>(h, hfinal);
}
